// EncoderSODA_63814624084601
// MI455X (gfx1250) — matrix operand order repaired and checked by reading, not compiled, not run on hardware
//
#include <hip/hip_runtime.h>
#include <hip/hip_bf16.h>

typedef __attribute__((ext_vector_type(16))) __bf16 v16bf;
typedef __attribute__((ext_vector_type(8)))  float  v8f;
typedef __attribute__((ext_vector_type(4)))  unsigned u32x4;

#define INPUT  512
#define HIDDEN 1024
#define ODIM   256
#define NSEQ   64
#define TLEN   512

__device__ inline unsigned short f2bf(float f) {
    unsigned u = __float_as_uint(f);
    u += 0x7FFFu + ((u >> 16) & 1u);           // round-to-nearest-even
    return (unsigned short)(u >> 16);
}
__device__ inline float sigf(float x) { return 1.0f / (1.0f + __expf(-x)); }

__device__ inline v8f wmma_bf16(u32x4 a0, u32x4 a1, u32x4 b0, u32x4 b1, v8f c) {
    union { u32x4 u[2]; v16bf v; } A, B;
    A.u[0] = a0; A.u[1] = a1; B.u[0] = b0; B.u[1] = b1;
    return __builtin_amdgcn_wmma_f32_16x16x32_bf16(
        /*neg_a=*/false, A.v, /*neg_b=*/false, B.v,
        /*c_mod=*/(short)0, c, /*reuse_a=*/false, /*reuse_b=*/false);
}

// ---------------- prep kernels ----------------

// bias = b_ih + b_hh; h_ping[0] = bf16(h0 * (1 - es[:,0])); bar = 0
__global__ void k_init(const float* __restrict__ h0, const float* __restrict__ es,
                       const float* __restrict__ b_ih, const float* __restrict__ b_hh,
                       float* __restrict__ bias, unsigned short* __restrict__ h0_bf,
                       int* __restrict__ bar) {
    int idx = blockIdx.x * 256 + threadIdx.x;
    if (idx < NSEQ * HIDDEN) {
        int m = idx / HIDDEN;
        float msk = 1.0f - es[m * TLEN + 0];
        h0_bf[idx] = f2bf(h0[idx] * msk);
    }
    if (idx < 4 * HIDDEN) bias[idx] = b_ih[idx] + b_hh[idx];
    if (idx == 0) *bar = 0;
}

__global__ void k_cvt_obs(const float* __restrict__ src, unsigned short* __restrict__ dst, int n) {
    int i = blockIdx.x * 256 + threadIdx.x;
    if (i < n) dst[i] = f2bf(src[i]);
}

// Pre-swizzle weight into CDNA5 dense 16-bit B-fragment layout (32x16 tiles).
// dst dword index = ((ntile*KC + kc)*32 + lane)*8 + v
//   col n = ntile*16 + lane%16 ; k = kc*32 + (v/4)*16 + (lane/16)*8 + (v%4)*2  (pair k, k+1)
// kn==0: src is [N,K]  (B[k][n] = src[n*K+k])     -> W_ih, W_hh
// kn==1: src is [K,N]  (B[k][n] = src[k*Ncols+n]) -> W_proj
__global__ void k_swizzle(const float* __restrict__ src, unsigned* __restrict__ dst,
                          int Kdim, int KC, int kn, int Ncols, int total) {
    int idx = blockIdx.x * 256 + threadIdx.x;
    if (idx >= total) return;
    int v     = idx & 7;
    int lane  = (idx >> 3) & 31;
    int rest  = idx >> 8;
    int kc    = rest % KC;
    int ntile = rest / KC;
    int n = ntile * 16 + (lane & 15);
    int k = kc * 32 + (v >> 2) * 16 + (lane >> 4) * 8 + (v & 3) * 2;
    float lo, hi;
    if (kn == 0) { lo = src[n * Kdim + k];  hi = src[n * Kdim + k + 1]; }
    else         { lo = src[k * Ncols + n]; hi = src[(k + 1) * Ncols + n]; }
    dst[idx] = (unsigned)f2bf(lo) | ((unsigned)f2bf(hi) << 16);
}

// ---------------- persistent LSTM kernel ----------------
// 64 workgroups x 256 threads (8 waves). WG wg owns hidden columns wg*16..+15.
// Wave pair (2w, 2w+1) computes gate w: wave handles M rows mhalf*32..+31
// (2 WMMA tiles). c lives in registers per WG for all 512 steps.
// h ping-pongs in global bf16; device-wide barrier once per step.
__global__ __launch_bounds__(256) void k_lstm(
    const unsigned short* __restrict__ obs_bf,
    const unsigned* __restrict__ WiF,
    const unsigned* __restrict__ WhF,
    const float* __restrict__ bias,
    const float* __restrict__ es,
    const float* __restrict__ c0,
    unsigned short* __restrict__ h_ping,
    unsigned short* __restrict__ hs_bf,
    int* __restrict__ bar) {
    const int wg    = blockIdx.x;          // 0..63
    const int tid   = threadIdx.x;
    const int wave  = tid >> 5;            // 0..7
    const int gate  = wave >> 1;           // 0..3 (i,f,g,o)
    const int mhalf = wave & 1;            // which 32 batch rows
    const int lane  = tid & 31;
    const int jcol  = lane & 15;
    const int khalf = lane >> 4;
    const int nt    = gate * 64 + wg;      // B n-tile in 4H gate space
    const int m0    = mhalf * 32;          // first batch row of this wave

    __shared__ float lds_g[4][NSEQ][16];

    float c_reg[4];
    int pm[4], pj[4];
#pragma unroll
    for (int r = 0; r < 4; r++) {
        int p = tid + 256 * r;             // 256*4 = 1024 (m,j) pairs
        pm[r] = p >> 4; pj[r] = p & 15;
        c_reg[r] = c0[pm[r] * HIDDEN + wg * 16 + pj[r]];
    }
    const float bias_n = bias[gate * HIDDEN + wg * 16 + jcol];

    for (int t = 0; t < TLEN; t++) {
        const unsigned short* hread  = h_ping + (t & 1) * (NSEQ * HIDDEN);
        unsigned short*       hwrite = h_ping + ((t + 1) & 1) * (NSEQ * HIDDEN);

        v8f acc[2];
#pragma unroll
        for (int mt = 0; mt < 2; mt++)
#pragma unroll
            for (int e = 0; e < 8; e++) acc[mt][e] = bias_n;

        // gates += x_t @ Wi^T   (K = 512)
        const unsigned* bi = WiF + (size_t)(nt * 16) * 256 + lane * 8;
#pragma unroll 2
        for (int kc = 0; kc < 16; kc++) {
            // request all fragments first, then issue WMMAs (overlap loads)
            u32x4 b0 = *(const u32x4*)(bi);
            u32x4 b1 = *(const u32x4*)(bi + 4);
            bi += 256;
            u32x4 a0[2], a1[2];
#pragma unroll
            for (int mt = 0; mt < 2; mt++) {
                const unsigned short* ar =
                    obs_bf + ((size_t)(m0 + mt * 16 + jcol) * TLEN + t) * INPUT + kc * 32 + khalf * 8;
                a0[mt] = *(const u32x4*)(ar);
                a1[mt] = *(const u32x4*)(ar + 16);
            }
#pragma unroll
            for (int mt = 0; mt < 2; mt++)
                acc[mt] = wmma_bf16(a0[mt], a1[mt], b0, b1, acc[mt]);
        }
        // gates += h @ Wh^T     (K = 1024)
        const unsigned* bh = WhF + (size_t)(nt * 32) * 256 + lane * 8;
#pragma unroll 2
        for (int kc = 0; kc < 32; kc++) {
            u32x4 b0 = *(const u32x4*)(bh);
            u32x4 b1 = *(const u32x4*)(bh + 4);
            bh += 256;
            u32x4 a0[2], a1[2];
#pragma unroll
            for (int mt = 0; mt < 2; mt++) {
                const unsigned short* ar =
                    hread + (size_t)(m0 + mt * 16 + jcol) * HIDDEN + kc * 32 + khalf * 8;
                a0[mt] = *(const u32x4*)(ar);
                a1[mt] = *(const u32x4*)(ar + 16);
            }
#pragma unroll
            for (int mt = 0; mt < 2; mt++)
                acc[mt] = wmma_bf16(a0[mt], a1[mt], b0, b1, acc[mt]);
        }

        __syncthreads();
#pragma unroll
        for (int mt = 0; mt < 2; mt++)
#pragma unroll
            for (int r = 0; r < 8; r++)
                lds_g[gate][m0 + mt * 16 + khalf * 8 + r][jcol] = acc[mt][r];
        __syncthreads();

        // elementwise cell update (c in registers)
#pragma unroll
        for (int r = 0; r < 4; r++) {
            int m = pm[r], j = pj[r];
            float gi = lds_g[0][m][j], gf = lds_g[1][m][j];
            float gg = lds_g[2][m][j], go = lds_g[3][m][j];
            float cm = c_reg[r] * (1.0f - es[m * TLEN + t]);
            float c  = sigf(gf) * cm + sigf(gi) * tanhf(gg);
            float h  = sigf(go) * tanhf(c);
            c_reg[r] = c;
            int col = wg * 16 + j;
            hs_bf[((size_t)m * TLEN + t) * HIDDEN + col] = f2bf(h);   // unmasked
            if (t + 1 < TLEN)                                         // mask for next carry
                hwrite[m * HIDDEN + col] = f2bf(h * (1.0f - es[m * TLEN + t + 1]));
        }

        // device-wide barrier (monotonic counter; zeroed per launch by k_init)
        __syncthreads();
        __threadfence();
        if (tid == 0) {
            atomicAdd(bar, 1);
            int target = 64 * (t + 1);
            volatile int* vb = bar;
            while (*vb < target) { __builtin_amdgcn_s_sleep(2); }
        }
        __syncthreads();
        __threadfence();
    }
}

// ---------------- projection GEMM: out = hs @ W_proj + b_proj ----------------
__global__ __launch_bounds__(256) void k_proj(
    const unsigned short* __restrict__ hs_bf,
    const unsigned* __restrict__ WpF,
    const float* __restrict__ b_proj,
    float* __restrict__ out) {
    const int tid   = threadIdx.x;
    const int lane  = tid & 31;
    const int wid   = blockIdx.x * 8 + (tid >> 5);  // 0..32767 tiles
    const int mtile = wid >> 4;                     // 0..2047
    const int ntile = wid & 15;                     // 0..15
    const int jcol  = lane & 15;
    const int khalf = lane >> 4;

    v8f acc;
    float b = b_proj[ntile * 16 + jcol];
#pragma unroll
    for (int e = 0; e < 8; e++) acc[e] = b;

    const unsigned* bp = WpF + (size_t)(ntile * 32) * 256 + lane * 8;
#pragma unroll 4
    for (int kc = 0; kc < 32; kc++) {
        u32x4 b0 = *(const u32x4*)(bp);
        u32x4 b1 = *(const u32x4*)(bp + 4);
        bp += 256;
        const unsigned short* ar =
            hs_bf + (size_t)(mtile * 16 + jcol) * HIDDEN + kc * 32 + khalf * 8;
        u32x4 a0 = *(const u32x4*)(ar);
        u32x4 a1 = *(const u32x4*)(ar + 16);
        acc = wmma_bf16(a0, a1, b0, b1, acc);
    }
#pragma unroll
    for (int r = 0; r < 8; r++) {
        int row = mtile * 16 + khalf * 8 + r;
        out[(size_t)row * ODIM + ntile * 16 + jcol] = acc[r];
    }
}

extern "C" void kernel_launch(void* const* d_in, const int* in_sizes, int n_in,
                              void* d_out, int out_size, void* d_ws, size_t ws_size,
                              hipStream_t stream) {
    const float* obs  = (const float*)d_in[0];
    const float* h0   = (const float*)d_in[1];
    const float* c0   = (const float*)d_in[2];
    const float* es   = (const float*)d_in[3];
    const float* W_ih = (const float*)d_in[4];
    const float* W_hh = (const float*)d_in[5];
    const float* b_ih = (const float*)d_in[6];
    const float* b_hh = (const float*)d_in[7];
    const float* W_pr = (const float*)d_in[8];
    const float* b_pr = (const float*)d_in[9];

    char* w = (char*)d_ws;
    size_t off = 0;
    unsigned short* obs_bf = (unsigned short*)(w + off); off += (size_t)NSEQ * TLEN * INPUT * 2;
    unsigned* WiF = (unsigned*)(w + off); off += (size_t)256 * 16 * 32 * 8 * 4;
    unsigned* WhF = (unsigned*)(w + off); off += (size_t)256 * 32 * 32 * 8 * 4;
    unsigned* WpF = (unsigned*)(w + off); off += (size_t)16 * 32 * 32 * 8 * 4;
    float* bias = (float*)(w + off); off += (size_t)4 * HIDDEN * 4;
    unsigned short* h_ping = (unsigned short*)(w + off); off += (size_t)2 * NSEQ * HIDDEN * 2;
    unsigned short* hs_bf  = (unsigned short*)(w + off); off += (size_t)NSEQ * TLEN * HIDDEN * 2;
    int* bar = (int*)(w + off);

    k_init<<<256, 256, 0, stream>>>(h0, es, b_ih, b_hh, bias, h_ping, bar);
    int nobs = NSEQ * TLEN * INPUT;
    k_cvt_obs<<<(nobs + 255) / 256, 256, 0, stream>>>(obs, obs_bf, nobs);
    k_swizzle<<<(256 * 16 * 256 + 255) / 256, 256, 0, stream>>>(W_ih, WiF, INPUT,  16, 0, 0,    256 * 16 * 256);
    k_swizzle<<<(256 * 32 * 256 + 255) / 256, 256, 0, stream>>>(W_hh, WhF, HIDDEN, 32, 0, 0,    256 * 32 * 256);
    k_swizzle<<<(16  * 32 * 256 + 255) / 256, 256, 0, stream>>>(W_pr, WpF, HIDDEN, 32, 1, ODIM, 16 * 32 * 256);
    k_lstm<<<64, 256, 0, stream>>>(obs_bf, WiF, WhF, bias, es, c0, h_ping, hs_bf, bar);
    k_proj<<<4096, 256, 0, stream>>>(hs_bf, WpF, b_pr, (float*)d_out);
}
